// GCN_11828339933416
// MI455X (gfx1250) — compile-verified, matrix operand order checked by reading, not run on hardware
//
#include <hip/hip_runtime.h>
#include <hip/hip_bf16.h>
#include <stdint.h>

// ---------------------------------------------------------------------------
// Types for WMMA / TDM (CDNA5 / gfx1250, wave32)
// ---------------------------------------------------------------------------
typedef __attribute__((ext_vector_type(16))) __bf16   v16bf;
typedef __attribute__((ext_vector_type(8)))  float    v8f;
typedef __attribute__((ext_vector_type(4)))  uint32_t v4u;
typedef __attribute__((ext_vector_type(8)))  int      v8i;
typedef __attribute__((ext_vector_type(4)))  int      v4i;

union BF16x16 {
    v16bf    v;
    uint32_t u[8];
};

__device__ __forceinline__ unsigned short f2bf(float f) {
    // round-to-nearest-even f32 -> bf16
    uint32_t u = __float_as_uint(f);
    uint32_t r = u + 0x7FFFu + ((u >> 16) & 1u);
    return (unsigned short)(r >> 16);
}

// ---------------------------------------------------------------------------
// Small elementwise kernels
// ---------------------------------------------------------------------------
__global__ void zero_f32_kernel(float* __restrict__ p, int n) {
    int i = blockIdx.x * blockDim.x + threadIdx.x;
    if (i < n) p[i] = 0.0f;
}

__global__ void degree_kernel(const int* __restrict__ dst, float* __restrict__ deg, int E) {
    int i = blockIdx.x * blockDim.x + threadIdx.x;
    if (i < E) atomicAdd(&deg[dst[i]], 1.0f);
}

__global__ void dinv_kernel(const float* __restrict__ deg, float* __restrict__ dinv, int n) {
    int i = blockIdx.x * blockDim.x + threadIdx.x;
    if (i < n) dinv[i] = rsqrtf(deg[i] + 1.0f);
}

// Convert W [K_REAL x N] (row-major, f32) -> Wt [N x K_PAD] (bf16, zero padded K)
__global__ void convert_wt_kernel(const float* __restrict__ W, unsigned short* __restrict__ Wt,
                                  int K_REAL, int K_PAD, int N) {
    int i = blockIdx.x * blockDim.x + threadIdx.x;
    if (i >= N * K_PAD) return;
    int c = i / K_PAD;
    int k = i - c * K_PAD;
    float v = (k < K_REAL) ? W[(size_t)k * N + c] : 0.0f;
    Wt[i] = f2bf(v);
}

// agg[i*F + f] = h[i*F + f] * dinv[i]^2   (self-loop contribution, full overwrite)
__global__ void selfloop_init_kernel(const float* __restrict__ h, const float* __restrict__ dinv,
                                     float* __restrict__ agg, int n, int F) {
    int i = blockIdx.x * blockDim.x + threadIdx.x;
    if (i >= n * F) return;
    int node = i / F;
    float di = dinv[node];
    agg[i] = h[i] * di * di;
}

template <int F, bool RELU>
__global__ void bias_act_kernel(float* __restrict__ agg, const float* __restrict__ b, int total) {
    int i = blockIdx.x * blockDim.x + threadIdx.x;
    if (i >= total) return;
    float v = agg[i] + b[i & (F - 1)];
    agg[i] = RELU ? fmaxf(v, 0.0f) : v;
}

// h_bf16[i] = bf16(relu(agg[i] + b[f]))  -- layer-1 activation, feeds GEMM2's A only
template <int F>
__global__ void bias_relu_bf16_kernel(const float* __restrict__ agg, const float* __restrict__ b,
                                      unsigned short* __restrict__ hbf, int total) {
    int i = blockIdx.x * blockDim.x + threadIdx.x;
    if (i >= total) return;
    float v = fmaxf(agg[i] + b[i & (F - 1)], 0.0f);
    hbf[i] = f2bf(v);
}

// ---------------------------------------------------------------------------
// Shared WMMA tile math: wave (wm, wn) owns a 16x16 C tile.
// A fragment (16x32 bf16) comes from LDS row-major [rows][K_PAD] bf16,
// B fragment from global transposed Wt [N][K_PAD] bf16.
// ---------------------------------------------------------------------------
template <int K_PAD, int N, int TM, int TN, bool GUARD>
__device__ __forceinline__ void wmma_tile_compute(const unsigned short* ldsA,
                                                  const unsigned short* __restrict__ Wt,
                                                  float* __restrict__ C, int M, int row0) {
    const int tid  = threadIdx.x;
    const int wave = tid >> 5;
    const int lane = tid & 31;
    const int wm   = wave / TN;
    const int wn   = wave - wm * TN;
    const int half = lane >> 4;
    const int l16  = lane & 15;

    const int r = wm * 16 + l16;  // A row in LDS tile (M = lane%16)
    const int c = wn * 16 + l16;  // B column         (N = lane%16)

    const uint32_t* ldsA32 = (const uint32_t*)ldsA;
    const uint32_t* wt32   = (const uint32_t*)Wt;

    v8f acc = {};
    for (int kc = 0; kc < K_PAD; kc += 32) {
        BF16x16 afr, bfr;
#pragma unroll
        for (int p = 0; p < 8; ++p) {
            // ISA 7.12.2 16-bit A 16x32 layout: p<4: K=8*half+2p ; p>=4: K=16+8*half+2(p-4)
            int k0 = kc + ((p < 4) ? (half * 8 + 2 * p) : (16 + half * 8 + 2 * (p - 4)));
            afr.u[p] = ldsA32[r * (K_PAD >> 1) + (k0 >> 1)];
            bfr.u[p] = wt32[c * (K_PAD >> 1) + (k0 >> 1)];
        }
        acc = __builtin_amdgcn_wmma_f32_16x16x32_bf16(
            false, afr.v, false, bfr.v, (short)0, acc, false, false);
    }

    // C/D layout: VGPR v -> M = v + 8*half, N = lane%16
#pragma unroll
    for (int v = 0; v < 8; ++v) {
        int mg = row0 + wm * 16 + v + half * 8;
        if (!GUARD || mg < M) C[(size_t)mg * N + (wn * 16 + l16)] = acc[v];
    }
}

// ---------------------------------------------------------------------------
// GEMM layer 1: A is f32 (x), converted f32->bf16 into LDS cooperatively.
// ---------------------------------------------------------------------------
template <int K_REAL, int K_PAD, int N, int TM, int TN, bool GUARD>
__global__ void wmma_gemm_f32a_kernel(const float* __restrict__ A,
                                      const unsigned short* __restrict__ Wt,
                                      float* __restrict__ C, int M) {
    __shared__ unsigned short ldsA[TM * 16 * K_PAD];
    const int tid  = threadIdx.x;
    const int row0 = blockIdx.x * (TM * 16);

    const int total = TM * 16 * K_PAD;
    for (int idx = tid; idx < total; idx += blockDim.x) {
        int r  = idx / K_PAD;
        int k  = idx - r * K_PAD;
        int rr = row0 + r;
        float v = 0.0f;
        if ((!GUARD || rr < M) && k < K_REAL) v = A[(size_t)rr * K_REAL + k];
        ldsA[idx] = f2bf(v);
    }
    __syncthreads();
    wmma_tile_compute<K_PAD, N, TM, TN, GUARD>(ldsA, Wt, C, M, row0);
}

// ---------------------------------------------------------------------------
// GEMM layer 2: A is already bf16 row-major [M][K]; the 2-D A tile is DMA'd
// into LDS by the Tensor Data Mover (tensor_load_to_lds + s_wait_tensorcnt).
// ---------------------------------------------------------------------------
template <int K_PAD, int N, int TM, int TN, bool GUARD>
__global__ void wmma_gemm_bf16a_tdm_kernel(const unsigned short* __restrict__ Abf,
                                           const unsigned short* __restrict__ Wt,
                                           float* __restrict__ C, int M) {
    __shared__ unsigned short ldsA[TM * 16 * K_PAD];
    const int row0 = blockIdx.x * (TM * 16);
    const int rows = GUARD ? min(TM * 16, M - row0) : (TM * 16);

#if __has_builtin(__builtin_amdgcn_tensor_load_to_lds) && __has_builtin(__builtin_amdgcn_s_wait_tensorcnt)
    if (threadIdx.x < 32) {  // one wave issues the TDM transfer
        uint64_t gaddr = (uint64_t)(const void*)(Abf + (size_t)row0 * K_PAD);
        uint32_t laddr =
            (uint32_t)(size_t)(__attribute__((address_space(3))) unsigned short*)ldsA;

        // D# group 0 (ISA 8.3): count=1 | lds_addr | global_addr[56:0] | type=2
        v4u g0;
        g0[0] = 1u;
        g0[1] = laddr;
        g0[2] = (uint32_t)gaddr;
        g0[3] = (uint32_t)((gaddr >> 32) & 0x01FFFFFFu) | (2u << 30);

        // D# group 1 (ISA 8.4): data_size=1 (2B); tensor_dim0=K_PAD; tensor_dim1=rows;
        // tile_dim0=K_PAD; tile_dim1=rows; tensor_dim0_stride=K_PAD.
        v8i g1;
        g1[0] = 1 << 16;                        // data_size=2 bytes
        g1[1] = (int)((uint32_t)(K_PAD & 0xFFFF) << 16);   // tensor_dim0[15:0]
        g1[2] = (int)((uint32_t)(rows & 0xFFFF) << 16);    // dim0 hi=0 | tensor_dim1[15:0]
        g1[3] = (int)((uint32_t)(K_PAD & 0xFFFF) << 16);   // dim1 hi=0 | tile_dim0
        g1[4] = rows;                           // tile_dim1 | tile_dim2=0
        g1[5] = K_PAD;                          // tensor_dim0_stride[31:0]
        g1[6] = 0;
        g1[7] = 0;

        v4i gz = {};
#if defined(__clang_major__) && (__clang_major__ >= 23)
        v8i gz8 = {};
        __builtin_amdgcn_tensor_load_to_lds(g0, g1, gz, gz, gz8, 0);
#else
        __builtin_amdgcn_tensor_load_to_lds(g0, g1, gz, gz, 0);
#endif
        __builtin_amdgcn_s_wait_tensorcnt(0);
    }
#else
    // Fallback: cooperative copy (bf16 already, no conversion needed)
    for (int idx = threadIdx.x; idx < rows * K_PAD; idx += blockDim.x)
        ldsA[idx] = Abf[(size_t)row0 * K_PAD + idx];
#endif
    __syncthreads();
    wmma_tile_compute<K_PAD, N, TM, TN, GUARD>(ldsA, Wt, C, M, row0);
}

// ---------------------------------------------------------------------------
// Edge scatter: wave-per-edge. agg[dst] += h[src] * dinv[src]*dinv[dst]
// (h and agg are L2-resident: 51.2MB / 25.6MB << 192MB L2)
// ---------------------------------------------------------------------------
template <int F>
__global__ void scatter_kernel(const float* __restrict__ h,
                               const int* __restrict__ src,
                               const int* __restrict__ dst,
                               const float* __restrict__ dinv,
                               float* __restrict__ agg, int E) {
    int wave = (blockIdx.x * blockDim.x + threadIdx.x) >> 5;
    int lane = threadIdx.x & 31;
    if (wave >= E) return;
    int s = src[wave];
    int d = dst[wave];
    float norm = dinv[s] * dinv[d];
    constexpr int PER = F / 32;
    const float* hs = h   + (size_t)s * F + lane * PER;
    float*       ad = agg + (size_t)d * F + lane * PER;
#pragma unroll
    for (int j = 0; j < PER; ++j) atomicAdd(&ad[j], hs[j] * norm);
}

// ---------------------------------------------------------------------------
// Decode: out[i] = dot(z[ea[i]], z[eb[i]]) over 64 features
// ---------------------------------------------------------------------------
__global__ void decode_kernel(const float* __restrict__ z,
                              const int* __restrict__ ea,
                              const int* __restrict__ eb,
                              float* __restrict__ out, int EL) {
    int i = blockIdx.x * blockDim.x + threadIdx.x;
    if (i >= EL) return;
    const float* za = z + (size_t)ea[i] * 64;
    const float* zb = z + (size_t)eb[i] * 64;
    float s = 0.0f;
#pragma unroll
    for (int k = 0; k < 64; ++k) s += za[k] * zb[k];
    out[i] = s;
}

// ---------------------------------------------------------------------------
// Host launcher
// ---------------------------------------------------------------------------
extern "C" void kernel_launch(void* const* d_in, const int* in_sizes, int n_in,
                              void* d_out, int out_size, void* d_ws, size_t ws_size,
                              hipStream_t stream) {
    constexpr int IN1 = 500;
    constexpr int K1P = 512;
    constexpr int HID = 128;
    constexpr int OUT = 64;

    const float* x   = (const float*)d_in[0];
    const int*   ei  = (const int*)d_in[1];
    const int*   eli = (const int*)d_in[2];
    const float* W1  = (const float*)d_in[3];
    const float* b1  = (const float*)d_in[4];
    const float* W2  = (const float*)d_in[5];
    const float* b2  = (const float*)d_in[6];
    float*       out = (float*)d_out;

    const int n  = in_sizes[0] / IN1;
    const int E  = in_sizes[1] / 2;
    const int EL = in_sizes[2] / 2;

    const int* src = ei;
    const int* dst = ei + E;
    const int* ela = eli;
    const int* elb = eli + EL;

    uint8_t* ws = (uint8_t*)d_ws;
    size_t off = 0;
    auto carve = [&](size_t bytes) -> uint8_t* {
        uint8_t* p = ws + off;
        off = (off + bytes + 255) & ~(size_t)255;
        return p;
    };
    float*          deg  = (float*)carve((size_t)n * 4);
    float*          dinv = (float*)carve((size_t)n * 4);
    unsigned short* Wt1  = (unsigned short*)carve((size_t)HID * K1P * 2);
    unsigned short* Wt2  = (unsigned short*)carve((size_t)OUT * HID * 2);
    float*          h1   = (float*)carve((size_t)n * HID * 4);   // layer1 pre-agg; later h2
    float*          agg1 = (float*)carve((size_t)n * HID * 4);
    unsigned short* hbf  = (unsigned short*)carve((size_t)n * HID * 2);  // relu(h) in bf16
    float*          agg2 = (float*)carve((size_t)n * OUT * 4);   // later z
    (void)ws_size;

    const int T = 256;
    auto blks = [](long long work, int thr) { return (unsigned)((work + thr - 1) / thr); };

    // 1) degrees + symmetric norm
    zero_f32_kernel<<<blks(n, T), T, 0, stream>>>(deg, n);
    degree_kernel<<<blks(E, T), T, 0, stream>>>(dst, deg, E);
    dinv_kernel<<<blks(n, T), T, 0, stream>>>(deg, dinv, n);

    // 2) weights -> transposed, zero-padded bf16
    convert_wt_kernel<<<blks((long long)HID * K1P, T), T, 0, stream>>>(W1, Wt1, IN1, K1P, HID);
    convert_wt_kernel<<<blks((long long)OUT * HID, T), T, 0, stream>>>(W2, Wt2, HID, HID, OUT);

    // 3) layer 1
    if (n % 16 == 0)
        wmma_gemm_f32a_kernel<IN1, K1P, HID, 1, 8, false><<<n / 16, 256, 0, stream>>>(x, Wt1, h1, n);
    else
        wmma_gemm_f32a_kernel<IN1, K1P, HID, 1, 8, true><<<blks(n, 16), 256, 0, stream>>>(x, Wt1, h1, n);
    selfloop_init_kernel<<<blks((long long)n * HID, T), T, 0, stream>>>(h1, dinv, agg1, n, HID);
    scatter_kernel<HID><<<blks((long long)E * 32, T), T, 0, stream>>>(h1, src, dst, dinv, agg1, E);
    bias_relu_bf16_kernel<HID><<<blks((long long)n * HID, T), T, 0, stream>>>(agg1, b1, hbf, n * HID);

    // 4) layer 2 (A tile via Tensor Data Mover; h2 reuses h1 buffer)
    if (n % 32 == 0)
        wmma_gemm_bf16a_tdm_kernel<HID, OUT, 2, 4, false><<<n / 32, 256, 0, stream>>>(hbf, Wt2, h1, n);
    else
        wmma_gemm_bf16a_tdm_kernel<HID, OUT, 2, 4, true><<<blks(n, 32), 256, 0, stream>>>(hbf, Wt2, h1, n);
    selfloop_init_kernel<<<blks((long long)n * OUT, T), T, 0, stream>>>(h1, dinv, agg2, n, OUT);
    scatter_kernel<OUT><<<blks((long long)E * 32, T), T, 0, stream>>>(h1, src, dst, dinv, agg2, E);
    bias_act_kernel<OUT, false><<<blks((long long)n * OUT, T), T, 0, stream>>>(agg2, b2, n * OUT);

    // 5) decode
    decode_kernel<<<blks(EL, T), T, 0, stream>>>(agg2, ela, elb, out, EL);
    (void)out_size; (void)n_in;
}
